// ModelSglang_38929583571306
// MI455X (gfx1250) — matrix operand order repaired and checked by reading, not compiled, not run on hardware
//
#include <hip/hip_runtime.h>
#include <stdint.h>

typedef __attribute__((ext_vector_type(16))) _Float16 v16h;
typedef __attribute__((ext_vector_type(8)))  float    v8f;

#define B_ 8
#define S_ 1024
#define H_ 16
#define D_ 64
#define TT (B_ * S_)       // total packed tokens = 8192
#define WAVES 4            // 4 waves/WG, each owns a 16-row Q tile
#define KP 72              // LDS row stride (f16) for K tile  (144B, 16B-mult, conflict-free)
#define VP 40              // LDS row stride (f16) for V^T tile ( 80B, 16B-mult, conflict-free)
#define PP 36              // LDS row stride (f32) for P tile   (144B, 16B-mult, conflict-free)

__device__ __forceinline__ uint32_t pack2h(float a, float b) {
    union { _Float16 h[2]; uint32_t u; } x;
    x.h[0] = (_Float16)a; x.h[1] = (_Float16)b;
    return x.u;   // lowers to v_cvt_pk_f16_f32
}

__device__ __forceinline__ v16h load_v16h_lds(const _Float16* p) {
    // two 16B LDS loads (avoid over-aligned 32B v16h load on 16B-aligned addr)
    union { uint4 q[2]; v16h v; } u;
    u.q[0] = *(const uint4*)p;
    u.q[1] = *(const uint4*)(p + 8);
    return u.v;
}

__device__ __forceinline__ void cvt4(v16h& d, int base, float4 f, float s) {
    d[base + 0] = (_Float16)(f.x * s);
    d[base + 1] = (_Float16)(f.y * s);
    d[base + 2] = (_Float16)(f.z * s);
    d[base + 3] = (_Float16)(f.w * s);
}

// ---------------- Prepass 1: K f32 -> f16 (same [T][H][D] layout) ----------------
__global__ __launch_bounds__(256)
void conv_k_kernel(const float* __restrict__ k, _Float16* __restrict__ k16)
{
    const size_t i = (size_t)blockIdx.x * 256 + threadIdx.x;   // i < T*H*D/4
    float4 f = ((const float4*)k)[i];
    uint2 o; o.x = pack2h(f.x, f.y); o.y = pack2h(f.z, f.w);
    ((uint2*)k16)[i] = o;
}

// ---------------- Prepass 2: V f32 [T][H][D] -> f16 V^T [H][D][T] ----------------
__global__ __launch_bounds__(256)
void transpose_v_kernel(const float* __restrict__ v, _Float16* __restrict__ vt)
{
    __shared__ float ldsT[D_][33];                 // [d][token-in-tile], padded
    const int tid = threadIdx.x;
    const int t0  = blockIdx.x * 32;               // token tile base
    const int h   = blockIdx.y;

    {   // coalesced read: row r (token), 8 consecutive d per thread
        const int r  = tid >> 3;
        const int cb = (tid & 7) * 8;
        const float4* src = (const float4*)(v + ((size_t)(t0 + r) * H_ + h) * D_ + cb);
        float4 a = src[0], b = src[1];
        ldsT[cb + 0][r] = a.x; ldsT[cb + 1][r] = a.y;
        ldsT[cb + 2][r] = a.z; ldsT[cb + 3][r] = a.w;
        ldsT[cb + 4][r] = b.x; ldsT[cb + 5][r] = b.y;
        ldsT[cb + 6][r] = b.z; ldsT[cb + 7][r] = b.w;
    }
    __syncthreads();
    {   // coalesced write: vt row d, 8 consecutive tokens per thread
        const int d  = tid >> 2;
        const int sb = (tid & 3) * 8;
        uint4 o;
        o.x = pack2h(ldsT[d][sb + 0], ldsT[d][sb + 1]);
        o.y = pack2h(ldsT[d][sb + 2], ldsT[d][sb + 3]);
        o.z = pack2h(ldsT[d][sb + 4], ldsT[d][sb + 5]);
        o.w = pack2h(ldsT[d][sb + 6], ldsT[d][sb + 7]);
        *(uint4*)(vt + (size_t)(h * D_ + d) * TT + t0 + sb) = o;
    }
}

// Stage one 32-key tile: K16 (32 x 128B) + V^T16 (64 x 64B) = 8KB via async DMA.
__device__ __forceinline__ void async_stage_tile16(const _Float16* k16head,
                                                   const _Float16* vthead,
                                                   _Float16* dK, _Float16* dV,
                                                   int kv0, int tid)
{
#pragma unroll
    for (int c = 0; c < 2; ++c) {                  // K: 256 x 16B segments
        const int id  = tid + 128 * c;
        const int row = id >> 3, seg = id & 7;
        const _Float16* gk = k16head + (size_t)(kv0 + row) * (H_ * D_) + seg * 8;
        const uint32_t lk  = (uint32_t)(uintptr_t)(dK + row * KP + seg * 8);
        asm volatile("global_load_async_to_lds_b128 %0, %1, off"
                     :: "v"(lk), "v"(gk) : "memory");
    }
#pragma unroll
    for (int c = 0; c < 2; ++c) {                  // V^T: 256 x 16B segments
        const int id = tid + 128 * c;
        const int dr = id >> 2, seg = id & 3;
        const _Float16* gv = vthead + (size_t)dr * TT + kv0 + seg * 8;
        const uint32_t lv  = (uint32_t)(uintptr_t)(dV + dr * VP + seg * 8);
        asm volatile("global_load_async_to_lds_b128 %0, %1, off"
                     :: "v"(lv), "v"(gv) : "memory");
    }
}

// ---------------- Main attention kernel (f16 K / V^T from workspace) ----------------
__global__ __launch_bounds__(WAVES * 32)
void fa_fwd_kernel16(const float* __restrict__ q,
                     const _Float16* __restrict__ k16,
                     const _Float16* __restrict__ vt,
                     const int* __restrict__ b_start_loc,
                     const int* __restrict__ b_seq_len,
                     float* __restrict__ out)
{
    __shared__ _Float16 lds_k16[2][32 * KP];   // double-buffered K tile (f16)
    __shared__ _Float16 lds_vt [2][D_ * VP];   // double-buffered V^T tile (f16)
    __shared__ float    lds_p[WAVES][16][PP];  // per-wave P transpose staging

    const int tid  = threadIdx.x;
    const int lane = tid & 31;
    const int wave = tid >> 5;
    const int lh   = lane >> 4;
    const int ln   = lane & 15;

    const int bh = blockIdx.y;
    const int b  = bh / H_;
    const int h  = bh % H_;
    const int seq_base = b_start_loc[b];
    const int seq_len  = b_seq_len[b];

    const int blk_q0 = blockIdx.x * (WAVES * 16);
    const int q0     = blk_q0 + wave * 16;

    const _Float16* k16head = k16 + ((size_t)seq_base * H_ + h) * D_;
    const _Float16* vthead  = vt + (size_t)(h * D_) * TT + seq_base;

    const float scale = 0.125f;

    // ---- Q tile into A layout (lane: m = ln; e -> d = 16lh+e) ----
    int qclamp = q0 + ln;
    qclamp = min(qclamp, seq_len - 1); qclamp = min(qclamp, S_ - 1); qclamp = max(qclamp, 0);
    const float4* q4 = (const float4*)(q + ((size_t)(seq_base + qclamp) * H_ + h) * D_);
    v16h aq0, aq1;
#pragma unroll
    for (int ch = 0; ch < 2; ++ch) {
        float4 f0 = q4[4 * lh + 0 + 8 * ch];
        float4 f1 = q4[4 * lh + 1 + 8 * ch];
        float4 f2 = q4[4 * lh + 2 + 8 * ch];
        float4 f3 = q4[4 * lh + 3 + 8 * ch];
        v16h& dst = ch ? aq1 : aq0;
        cvt4(dst, 0, f0, scale); cvt4(dst, 4, f1, scale);
        cvt4(dst, 8, f2, scale); cvt4(dst, 12, f3, scale);
    }

    v8f acc0 = {}, acc1 = {}, acc2 = {}, acc3 = {};
    float m_r[8], l_r[8];
#pragma unroll
    for (int e = 0; e < 8; ++e) { m_r[e] = -1e30f; l_r[e] = 0.0f; }

    const int kv_max    = min(blk_q0 + WAVES * 16, seq_len);
    const int nIter     = (kv_max > 0) ? (kv_max + 31) / 32 : 0;
    const int my_kv_end = (q0 < seq_len) ? min(q0 + 16, seq_len) : 0;

    if (nIter > 0)
        async_stage_tile16(k16head, vthead, lds_k16[0], lds_vt[0], 0, tid);
    asm volatile("s_wait_asynccnt 0x0" ::: "memory");
    __syncthreads();

    for (int it = 0; it < nIter; ++it) {
        const int kv0 = it * 32;
        const int cur = it & 1;

        if (it + 1 < nIter)
            async_stage_tile16(k16head, vthead, lds_k16[cur ^ 1], lds_vt[cur ^ 1],
                               kv0 + 32, tid);

        if (kv0 < my_kv_end) {     // wave-uniform
            const _Float16* Kt = lds_k16[cur];
            const _Float16* Vt = lds_vt[cur];

            // ---- Scores ----
            v8f s01[2];
#pragma unroll
            for (int j = 0; j < 2; ++j) {
                const _Float16* kr = Kt + (16 * j + ln) * KP + 16 * lh;
                v16h bk0 = load_v16h_lds(kr);        // d-chunk 0
                v16h bk1 = load_v16h_lds(kr + 32);   // d-chunk 1
                v8f c = {};
                c = __builtin_amdgcn_wmma_f32_16x16x32_f16(false, aq0, false, bk0,
                                                           (short)0, c, false, false);
                c = __builtin_amdgcn_wmma_f32_16x16x32_f16(false, aq1, false, bk1,
                                                           (short)0, c, false, false);
                s01[j] = c;
            }

            // ---- Online softmax (C layout: row m = 8lh+e, col n = ln) ----
            float alpha[8];
#pragma unroll
            for (int e = 0; e < 8; ++e) {
                const int qrow = q0 + (lh << 3) + e;
                const int key0 = kv0 + ln;
                const int key1 = kv0 + 16 + ln;
                float s0 = (key0 <= qrow && key0 < seq_len) ? s01[0][e] : -1e30f;
                float s1 = (key1 <= qrow && key1 < seq_len) ? s01[1][e] : -1e30f;

                float rm = fmaxf(s0, s1);
#pragma unroll
                for (int off = 8; off >= 1; off >>= 1)
                    rm = fmaxf(rm, __shfl_xor(rm, off, 32));

                const float mnew = fmaxf(m_r[e], rm);
                const float a    = __expf(m_r[e] - mnew);
                const float p0   = __expf(s0 - mnew);
                const float p1   = __expf(s1 - mnew);
                float rs = p0 + p1;
#pragma unroll
                for (int off = 8; off >= 1; off >>= 1)
                    rs += __shfl_xor(rs, off, 32);

                l_r[e]   = l_r[e] * a + rs;
                m_r[e]   = mnew;
                alpha[e] = a;

                lds_p[wave][(lh << 3) + e][ln]      = p0;
                lds_p[wave][(lh << 3) + e][16 + ln] = p1;
            }

#pragma unroll
            for (int e = 0; e < 8; ++e) {
                acc0[e] *= alpha[e]; acc1[e] *= alpha[e];
                acc2[e] *= alpha[e]; acc3[e] *= alpha[e];
            }

            // ---- P back into A layout: four contiguous float4 per lane ----
            const float4* pr = (const float4*)(&lds_p[wave][ln][0] + 16 * lh);
            v16h ap;
            cvt4(ap, 0,  pr[0], 1.0f);   // k = 16lh+0..3
            cvt4(ap, 4,  pr[1], 1.0f);   // k = 16lh+4..7
            cvt4(ap, 8,  pr[2], 1.0f);   // k = 16lh+8..11
            cvt4(ap, 12, pr[3], 1.0f);   // k = 16lh+12..15

            // ---- O += P * V : V^T rows are contiguous f16 now ----
#pragma unroll
            for (int c2 = 0; c2 < 4; ++c2) {
                v16h bv = load_v16h_lds(Vt + (16 * c2 + ln) * VP + 16 * lh);
                v8f* accp = (c2 == 0) ? &acc0 : (c2 == 1) ? &acc1
                          : (c2 == 2) ? &acc2 : &acc3;
                *accp = __builtin_amdgcn_wmma_f32_16x16x32_f16(false, ap, false, bv,
                                                               (short)0, *accp,
                                                               false, false);
            }
        }

        asm volatile("s_wait_asynccnt 0x0" ::: "memory");
        __syncthreads();
    }

#pragma unroll
    for (int e = 0; e < 8; ++e) {
        const int qrow = q0 + (lh << 3) + e;
        const float inv = (qrow < seq_len && l_r[e] > 0.0f) ? (1.0f / l_r[e]) : 0.0f;
        float* optr = out + ((size_t)(seq_base + qrow) * H_ + h) * D_;
        optr[ln]      = acc0[e] * inv;
        optr[16 + ln] = acc1[e] * inv;
        optr[32 + ln] = acc2[e] * inv;
        optr[48 + ln] = acc3[e] * inv;
    }
}

// ---------------- Fallback (round-2 f32 kernel) used if ws is too small ----------------
#define KSTRIDE 68
__device__ __forceinline__ void async_stage_tile_f32(const float* kghead, const float* vghead,
                                                     float* dK, float* dV,
                                                     int kv0, int seq_len, int tid)
{
#pragma unroll
    for (int c = 0; c < 4; ++c) {
        const int id  = tid + 128 * c;
        const int row = id >> 4, seg = id & 15;
        int key = kv0 + row;
        key = min(key, seq_len - 1); key = min(key, S_ - 1); key = max(key, 0);
        const float* gk = kghead + (size_t)key * (H_ * D_) + seg * 4;
        const float* gv = vghead + (size_t)key * (H_ * D_) + seg * 4;
        const uint32_t lk = (uint32_t)(uintptr_t)(dK + row * KSTRIDE + seg * 4);
        const uint32_t lv = (uint32_t)(uintptr_t)(dV + row * KSTRIDE + seg * 4);
        asm volatile("global_load_async_to_lds_b128 %0, %1, off" :: "v"(lk), "v"(gk) : "memory");
        asm volatile("global_load_async_to_lds_b128 %0, %1, off" :: "v"(lv), "v"(gv) : "memory");
    }
}

__global__ __launch_bounds__(WAVES * 32)
void fa_fwd_kernel_f32(const float* __restrict__ q, const float* __restrict__ k,
                       const float* __restrict__ v, const int* __restrict__ b_start_loc,
                       const int* __restrict__ b_seq_len, float* __restrict__ out)
{
    __shared__ float lds_k[2][32 * KSTRIDE];
    __shared__ float lds_v[2][32 * KSTRIDE];
    __shared__ float lds_p[WAVES][16][PP];

    const int tid = threadIdx.x, lane = tid & 31, wave = tid >> 5;
    const int lh = lane >> 4, ln = lane & 15;
    const int bh = blockIdx.y, b = bh / H_, h = bh % H_;
    const int seq_base = b_start_loc[b], seq_len = b_seq_len[b];
    const int blk_q0 = blockIdx.x * (WAVES * 16);
    const int q0 = blk_q0 + wave * 16;
    const float* kghead = k + ((size_t)seq_base * H_ + h) * D_;
    const float* vghead = v + ((size_t)seq_base * H_ + h) * D_;
    const float scale = 0.125f;

    int qclamp = q0 + ln;
    qclamp = min(qclamp, seq_len - 1); qclamp = min(qclamp, S_ - 1); qclamp = max(qclamp, 0);
    const float4* q4 = (const float4*)(q + ((size_t)(seq_base + qclamp) * H_ + h) * D_);
    v16h aq0, aq1;
#pragma unroll
    for (int ch = 0; ch < 2; ++ch) {
        float4 f0 = q4[4 * lh + 0 + 8 * ch], f1 = q4[4 * lh + 1 + 8 * ch];
        float4 f2 = q4[4 * lh + 2 + 8 * ch], f3 = q4[4 * lh + 3 + 8 * ch];
        v16h& dst = ch ? aq1 : aq0;
        cvt4(dst, 0, f0, scale); cvt4(dst, 4, f1, scale);
        cvt4(dst, 8, f2, scale); cvt4(dst, 12, f3, scale);
    }
    v8f acc0 = {}, acc1 = {}, acc2 = {}, acc3 = {};
    float m_r[8], l_r[8];
#pragma unroll
    for (int e = 0; e < 8; ++e) { m_r[e] = -1e30f; l_r[e] = 0.0f; }
    const int kv_max = min(blk_q0 + WAVES * 16, seq_len);
    const int nIter = (kv_max > 0) ? (kv_max + 31) / 32 : 0;
    const int my_kv_end = (q0 < seq_len) ? min(q0 + 16, seq_len) : 0;

    if (nIter > 0) async_stage_tile_f32(kghead, vghead, lds_k[0], lds_v[0], 0, seq_len, tid);
    asm volatile("s_wait_asynccnt 0x0" ::: "memory");
    __syncthreads();

    for (int it = 0; it < nIter; ++it) {
        const int kv0 = it * 32, cur = it & 1;
        if (it + 1 < nIter)
            async_stage_tile_f32(kghead, vghead, lds_k[cur ^ 1], lds_v[cur ^ 1],
                                 kv0 + 32, seq_len, tid);
        if (kv0 < my_kv_end) {
            const float* Kt = lds_k[cur];
            const float* Vt = lds_v[cur];
            v8f s01[2];
#pragma unroll
            for (int j = 0; j < 2; ++j) {
                const float4* kr = (const float4*)(Kt + (16 * j + ln) * KSTRIDE + 16 * lh);
                v16h bk0, bk1;
                cvt4(bk0, 0, kr[0], 1.0f);  cvt4(bk0, 4, kr[1], 1.0f);
                cvt4(bk0, 8, kr[2], 1.0f);  cvt4(bk0, 12, kr[3], 1.0f);
                cvt4(bk1, 0, kr[8], 1.0f);  cvt4(bk1, 4, kr[9], 1.0f);
                cvt4(bk1, 8, kr[10], 1.0f); cvt4(bk1, 12, kr[11], 1.0f);
                v8f c = {};
                c = __builtin_amdgcn_wmma_f32_16x16x32_f16(false, aq0, false, bk0, (short)0, c, false, false);
                c = __builtin_amdgcn_wmma_f32_16x16x32_f16(false, aq1, false, bk1, (short)0, c, false, false);
                s01[j] = c;
            }
            float alpha[8];
#pragma unroll
            for (int e = 0; e < 8; ++e) {
                const int qrow = q0 + (lh << 3) + e;
                const int key0 = kv0 + ln, key1 = kv0 + 16 + ln;
                float s0 = (key0 <= qrow && key0 < seq_len) ? s01[0][e] : -1e30f;
                float s1 = (key1 <= qrow && key1 < seq_len) ? s01[1][e] : -1e30f;
                float rm = fmaxf(s0, s1);
#pragma unroll
                for (int off = 8; off >= 1; off >>= 1)
                    rm = fmaxf(rm, __shfl_xor(rm, off, 32));
                const float mnew = fmaxf(m_r[e], rm);
                const float a = __expf(m_r[e] - mnew);
                const float p0 = __expf(s0 - mnew), p1 = __expf(s1 - mnew);
                float rs = p0 + p1;
#pragma unroll
                for (int off = 8; off >= 1; off >>= 1)
                    rs += __shfl_xor(rs, off, 32);
                l_r[e] = l_r[e] * a + rs; m_r[e] = mnew; alpha[e] = a;
                lds_p[wave][(lh << 3) + e][ln] = p0;
                lds_p[wave][(lh << 3) + e][16 + ln] = p1;
            }
#pragma unroll
            for (int e = 0; e < 8; ++e) {
                acc0[e] *= alpha[e]; acc1[e] *= alpha[e];
                acc2[e] *= alpha[e]; acc3[e] *= alpha[e];
            }
            const float4* pr = (const float4*)(&lds_p[wave][ln][0] + 16 * lh);
            v16h ap;
            cvt4(ap, 0, pr[0], 1.0f); cvt4(ap, 4, pr[1], 1.0f);
            cvt4(ap, 8, pr[2], 1.0f); cvt4(ap, 12, pr[3], 1.0f);
#pragma unroll
            for (int c2 = 0; c2 < 4; ++c2) {
                v16h bv;
#pragma unroll
                for (int e = 0; e < 16; ++e)
                    bv[e] = (_Float16)Vt[(16 * lh + e) * KSTRIDE + (c2 << 4) + ln];
                v8f* accp = (c2 == 0) ? &acc0 : (c2 == 1) ? &acc1 : (c2 == 2) ? &acc2 : &acc3;
                *accp = __builtin_amdgcn_wmma_f32_16x16x32_f16(false, ap, false, bv, (short)0, *accp, false, false);
            }
        }
        asm volatile("s_wait_asynccnt 0x0" ::: "memory");
        __syncthreads();
    }
#pragma unroll
    for (int e = 0; e < 8; ++e) {
        const int qrow = q0 + (lh << 3) + e;
        const float inv = (qrow < seq_len && l_r[e] > 0.0f) ? (1.0f / l_r[e]) : 0.0f;
        float* optr = out + ((size_t)(seq_base + qrow) * H_ + h) * D_;
        optr[ln] = acc0[e] * inv;      optr[16 + ln] = acc1[e] * inv;
        optr[32 + ln] = acc2[e] * inv; optr[48 + ln] = acc3[e] * inv;
    }
}

extern "C" void kernel_launch(void* const* d_in, const int* in_sizes, int n_in,
                              void* d_out, int out_size, void* d_ws, size_t ws_size,
                              hipStream_t stream) {
    (void)in_sizes; (void)n_in; (void)out_size;
    const float* q = (const float*)d_in[0];
    const float* k = (const float*)d_in[1];
    const float* v = (const float*)d_in[2];
    const int* b_start_loc = (const int*)d_in[3];
    const int* b_seq_len   = (const int*)d_in[4];
    float* out = (float*)d_out;

    const size_t elems = (size_t)TT * H_ * D_;        // 8.4M
    const size_t need  = elems * 2 * 2;               // k16 + vt, f16 each = 32MB

    dim3 grid(S_ / (WAVES * 16), B_ * H_);            // (16, 128)
    dim3 block(WAVES * 32);                           // 128 threads = 4 waves

    if (ws_size >= need) {
        _Float16* k16 = (_Float16*)d_ws;
        _Float16* vt  = k16 + elems;
        conv_k_kernel<<<(uint32_t)(elems / 4 / 256), 256, 0, stream>>>(k, k16);
        transpose_v_kernel<<<dim3(TT / 32, H_), 256, 0, stream>>>(v, vt);
        fa_fwd_kernel16<<<grid, block, 0, stream>>>(q, k16, vt, b_start_loc, b_seq_len, out);
    } else {
        fa_fwd_kernel_f32<<<grid, block, 0, stream>>>(q, k, v, b_start_loc, b_seq_len, out);
    }
}
